// Ours_49091476193595
// MI455X (gfx1250) — matrix operand order repaired and checked by reading, not compiled, not run on hardware
//
#include <hip/hip_runtime.h>
#include <hip/hip_bf16.h>
#include <math.h>

typedef __bf16 bf16;
typedef __attribute__((ext_vector_type(16))) bf16  v16bf;
typedef __attribute__((ext_vector_type(8)))  float v8f;

namespace vitp {

constexpr int D = 768, DEPTH = 12, HEADS = 12, HD = 64;
constexpr int P = 16, IMG = 224, GRD = 14, NPATCH = GRD * GRD;   // 196
constexpr int NTOK = NPATCH + 1;                                  // 197
constexpr int PLEN = 5, NTASK = 5, NCLS = 100;
constexpr int BATCH = 16;
constexpr int NSUB = NTOK + PLEN;                                 // 202
constexpr int KPAD = 224;   // 202 padded to 7*32 (K of AV gemm / cols of scores)
constexpr int MPAD = 208;   // 197 padded to 13*16 (row pitch of score buffer)
constexpr float SCALE = 0.125f;  // HD^-0.5

__device__ __forceinline__ float wsum(float x) {
  for (int o = 16; o; o >>= 1) x += __shfl_xor(x, o, 32);
  return x;
}
__device__ __forceinline__ float wmaxr(float x) {
  for (int o = 16; o; o >>= 1) x = fmaxf(x, __shfl_xor(x, o, 32));
  return x;
}

__device__ __forceinline__ void pack8(v16bf& dst, int base, float4 x, float4 y) {
  dst[base + 0] = (bf16)x.x; dst[base + 1] = (bf16)x.y;
  dst[base + 2] = (bf16)x.z; dst[base + 3] = (bf16)x.w;
  dst[base + 4] = (bf16)y.x; dst[base + 5] = (bf16)y.y;
  dst[base + 6] = (bf16)y.z; dst[base + 7] = (bf16)y.w;
}

// ---------------------------------------------------------------------------
// WMMA GEMM:  C[m,n] = sum_k A[m,k] * B[k,n] (+bias[n]) (+GELU)
// TRANSB==0: W is (N,K) row-major (x @ w.T, torch-style weights)
// TRANSB==1: W is (K,N) row-major (attention P @ V)
// Each wave owns a 16(M) x 32(N) output tile: one A fragment, two B fragments,
// two v_wmma per K-step. 4 waves/block walk 4 consecutive M tiles.
// Loads are branchless: OOB rows/cols are clamped for addressing; an OOB A row
// only feeds unstored output rows, an OOB B col only feeds unstored output
// cols, so stored results are exact. Requires K % 32 == 0 (true at all sites).
// ---------------------------------------------------------------------------
template <int TRANSB, int ACT>
__global__ __launch_bounds__(128) void k_gemm(
    const float* __restrict__ A, const float* __restrict__ W,
    const float* __restrict__ bias, float* __restrict__ C,
    int M, int N, int K, int lda, int ldw, int ldc,
    long long sA, long long sW, long long sC) {
  const int lane = threadIdx.x & 31;
  const int wid = threadIdx.x >> 5;
  const int tn = blockIdx.x;                 // 32-wide N tile
  const int tm = blockIdx.y * 4 + wid;       // 16-high M tile per wave
  const int bz = blockIdx.z;
  const float* Ab = A + sA * bz;
  const float* Wb = W + sW * bz;
  float*       Cb = C + sC * bz;

  const int g = lane >> 4, l = lane & 15;
  const int row = tm * 16 + l;
  const int rowl = row < M ? row : (M - 1);          // clamp for loads
  const int col0 = tn * 32 + l;
  const int col1 = col0 + 16;
  const int col0l = col0 < N ? col0 : (N - 1);
  const int col1l = col1 < N ? col1 : (N - 1);

  const float* ap = Ab + (long long)rowl * lda;
  const float* bp0 = TRANSB ? (Wb + col0l) : (Wb + (long long)col0l * ldw);
  const float* bp1 = TRANSB ? (Wb + col1l) : (Wb + (long long)col1l * ldw);

  v8f acc0 = {0.f, 0.f, 0.f, 0.f, 0.f, 0.f, 0.f, 0.f};
  v8f acc1 = {0.f, 0.f, 0.f, 0.f, 0.f, 0.f, 0.f, 0.f};

  for (int k0 = 0; k0 < K; k0 += 32) {
    // ---- A fragment: lane holds contiguous K run [k0+16g, +16) ----
    const float* r0 = ap + k0 + (g << 4);
    if (k0 + 32 < K) __builtin_prefetch(r0 + 32, 0, 1);
    const float4 A0 = *(const float4*)(r0);
    const float4 A1 = *(const float4*)(r0 + 4);
    const float4 A2 = *(const float4*)(r0 + 8);
    const float4 A3 = *(const float4*)(r0 + 12);
    v16bf a;
    pack8(a, 0, A0, A1);
    pack8(a, 8, A2, A3);

    v16bf b0v, b1v;
    if (TRANSB) {
      // B column strided: K = k0+16g+j, j=0..15
#pragma unroll
      for (int v = 0; v < 8; ++v) {
        const long long kk = (long long)(k0 + (g << 4) + 2 * v) * ldw;
        b0v[2 * v]     = (bf16)bp0[kk];
        b0v[2 * v + 1] = (bf16)bp0[kk + ldw];
        b1v[2 * v]     = (bf16)bp1[kk];
        b1v[2 * v + 1] = (bf16)bp1[kk + ldw];
      }
    } else {
      // B lane holds contiguous K run [k0+16g, +16) of its W row
      const float* q0 = bp0 + k0 + (g << 4);
      const float* q1 = bp1 + k0 + (g << 4);
      const float4 B0 = *(const float4*)(q0);
      const float4 B1 = *(const float4*)(q0 + 4);
      const float4 B2 = *(const float4*)(q0 + 8);
      const float4 B3 = *(const float4*)(q0 + 12);
      const float4 B4 = *(const float4*)(q1);
      const float4 B5 = *(const float4*)(q1 + 4);
      const float4 B6 = *(const float4*)(q1 + 8);
      const float4 B7 = *(const float4*)(q1 + 12);
      pack8(b0v, 0, B0, B1);
      pack8(b0v, 8, B2, B3);
      pack8(b1v, 0, B4, B5);
      pack8(b1v, 8, B6, B7);
    }
    acc0 = __builtin_amdgcn_wmma_f32_16x16x32_bf16(false, a, false, b0v,
                                                   (short)0, acc0, false, false);
    acc1 = __builtin_amdgcn_wmma_f32_16x16x32_bf16(false, a, false, b1v,
                                                   (short)0, acc1, false, false);
  }

  const float bv0 = (bias != nullptr && col0 < N) ? bias[col0] : 0.f;
  const float bv1 = (bias != nullptr && col1 < N) ? bias[col1] : 0.f;
#pragma unroll
  for (int r = 0; r < 8; ++r) {
    const int rm = tm * 16 + r + (g << 3);   // C: VGPR r -> row r (+8 hi half)
    if (rm < M) {
      float x0 = acc0[r] + bv0;
      float x1 = acc1[r] + bv1;
      if (ACT == 1) {
        x0 = 0.5f * x0 * (1.f + erff(x0 * 0.70710678118f));
        x1 = 0.5f * x1 * (1.f + erff(x1 * 0.70710678118f));
      }
      float* crow = Cb + (long long)rm * ldc;
      if (col0 < N) crow[col0] = x0;
      if (col1 < N) crow[col1] = x1;
    }
  }
}

// ------------------------------ LayerNorm ----------------------------------
__global__ void k_ln(const float* __restrict__ x, const float* __restrict__ w,
                     const float* __restrict__ b, float* __restrict__ y, int rows) {
  const int wid = blockIdx.x * (blockDim.x >> 5) + (threadIdx.x >> 5);
  if (wid >= rows) return;
  const int lane = threadIdx.x & 31;
  const float4* xr = (const float4*)(x + (long long)wid * D);
  float s = 0.f;
  for (int j = lane; j < D / 4; j += 32) {
    const float4 t = xr[j];
    s += t.x + t.y + t.z + t.w;
  }
  s = wsum(s);
  const float mean = s * (1.f / D);
  float v = 0.f;
  for (int j = lane; j < D / 4; j += 32) {
    const float4 t = xr[j];
    const float d0 = t.x - mean, d1 = t.y - mean, d2 = t.z - mean, d3 = t.w - mean;
    v += d0 * d0 + d1 * d1 + d2 * d2 + d3 * d3;
  }
  v = wsum(v);
  const float rstd = rsqrtf(v * (1.f / D) + 1e-6f);
  const float4* wv = (const float4*)w;
  const float4* bv = (const float4*)b;
  float4* yr = (float4*)(y + (long long)wid * D);
  for (int j = lane; j < D / 4; j += 32) {
    const float4 t = xr[j], ww = wv[j], bb = bv[j];
    float4 o;
    o.x = (t.x - mean) * rstd * ww.x + bb.x;
    o.y = (t.y - mean) * rstd * ww.y + bb.y;
    o.z = (t.z - mean) * rstd * ww.z + bb.z;
    o.w = (t.w - mean) * rstd * ww.w + bb.w;
    yr[j] = o;
  }
}

// ------------------------------ residual add (n % 4 == 0) ------------------
__global__ void k_add(float* __restrict__ y, const float* __restrict__ t, long long n4) {
  float4* y4 = (float4*)y;
  const float4* t4 = (const float4*)t;
  for (long long i = blockIdx.x * (long long)blockDim.x + threadIdx.x; i < n4;
       i += (long long)gridDim.x * blockDim.x) {
    float4 a = y4[i];
    const float4 b = t4[i];
    a.x += b.x; a.y += b.y; a.z += b.z; a.w += b.w;
    y4[i] = a;
  }
}

// --------------------- standard MHA, online softmax ------------------------
// qkv: (B, N, 3*D) rows; out: (B, N, D). One wave per (b, h, n); lane owns 2 dims.
__global__ __launch_bounds__(32) void k_mha(const float* __restrict__ qkv,
                                            float* __restrict__ out, int Ntok) {
  const int n = blockIdx.x, h = blockIdx.y, b = blockIdx.z;
  const int lane = threadIdx.x & 31;
  const float* base = qkv + (long long)b * Ntok * 3 * D;
  const float* qp = base + (long long)n * 3 * D + h * HD;
  const float q0 = qp[2 * lane], q1 = qp[2 * lane + 1];
  float mx = -1e30f, ssum = 0.f, o0 = 0.f, o1 = 0.f;
  for (int m = 0; m < Ntok; ++m) {
    const float* kp = base + (long long)m * 3 * D + D + h * HD;
    float p = q0 * kp[2 * lane] + q1 * kp[2 * lane + 1];
    p = wsum(p) * SCALE;
    const float nmx = fmaxf(mx, p);
    const float corr = __expf(mx - nmx);
    const float e = __expf(p - nmx);
    const float* vp = base + (long long)m * 3 * D + 2 * D + h * HD;
    ssum = ssum * corr + e;
    o0 = o0 * corr + e * vp[2 * lane];
    o1 = o1 * corr + e * vp[2 * lane + 1];
    mx = nmx;
  }
  const float inv = 1.f / ssum;
  float* op = out + ((long long)b * Ntok + n) * D + h * HD;
  op[2 * lane] = o0 * inv;
  op[2 * lane + 1] = o1 * inv;
}

// ------- deep block: build padded K/V = [prompt_k; k ; 0] / [prompt_v; v; 0] -
__global__ void k_buildkv(const float* __restrict__ qkv,
                          const float* __restrict__ prompts,
                          const int* __restrict__ selc,
                          float* __restrict__ Kp, float* __restrict__ Vp, int li) {
  const int r = blockIdx.x, b = blockIdx.y;
  float* kr = Kp + ((long long)b * KPAD + r) * D;
  float* vr = Vp + ((long long)b * KPAD + r) * D;
  const float *ks = nullptr, *vs = nullptr;
  if (r < PLEN) {
    const long long base = (long long)selc[b] * 2 * 3 * PLEN * D;  // (NTASK,1,2,3,PLEN,D)
    ks = prompts + base + (long long)(2 * li) * PLEN * D + (long long)r * D;
    vs = prompts + base + (long long)(2 * li + 1) * PLEN * D + (long long)r * D;
  } else if (r < PLEN + NTOK) {
    const float* q = qkv + ((long long)b * NTOK + (r - PLEN)) * 3 * D;
    ks = q + D;
    vs = q + 2 * D;
  }
  for (int j = threadIdx.x; j < D; j += blockDim.x) {
    kr[j] = ks ? ks[j] : 0.f;
    vr[j] = vs ? vs[j] : 0.f;
  }
}

// scaled softmax over valid 202 cols; zero the 22 pad cols so P@V is unaffected
__global__ __launch_bounds__(32) void k_softmax(float* __restrict__ S) {
  const int n = blockIdx.x, b = blockIdx.y, lane = threadIdx.x & 31;
  float* row = S + ((long long)b * MPAD + n) * KPAD;
  const int L = PLEN + NTOK;
  float m = -1e30f;
  for (int j = lane; j < L; j += 32) m = fmaxf(m, row[j] * SCALE);
  m = wmaxr(m);
  float s = 0.f;
  for (int j = lane; j < L; j += 32) {
    const float e = __expf(row[j] * SCALE - m);
    row[j] = e;
    s += e;
  }
  s = wsum(s);
  const float inv = 1.f / s;
  for (int j = lane; j < KPAD; j += 32) row[j] = (j < L) ? row[j] * inv : 0.f;
}

// av.swapaxes(1,2).reshape(B,N,D) remap (faithful to reference semantics)
__global__ void k_avremap(const float* __restrict__ AV, float* __restrict__ O) {
  const long long per = (long long)NTOK * D;
  const long long total = (long long)BATCH * per;
  for (long long i = blockIdx.x * (long long)blockDim.x + threadIdx.x; i < total;
       i += (long long)gridDim.x * blockDim.x) {
    const long long f = i % per;
    const int b = (int)(i / per);
    const int nsrc = (int)(f % NTOK);
    const int dsrc = (int)(f / NTOK);
    O[i] = AV[((long long)b * NTOK + nsrc) * D + dsrc];
  }
}

// ------------------------------ embedding ----------------------------------
__global__ void k_patch(const float* __restrict__ in, float* __restrict__ Xp) {
  const long long total = (long long)BATCH * NPATCH * D;
  for (long long i = blockIdx.x * (long long)blockDim.x + threadIdx.x; i < total;
       i += (long long)gridDim.x * blockDim.x) {
    const int cc = (int)(i % D);
    const int g = (int)((i / D) % NPATCH);
    const int b = (int)(i / ((long long)D * NPATCH));
    const int c = cc >> 8, py = (cc >> 4) & 15, px = cc & 15;
    const int gy = g / GRD, gx = g % GRD;
    Xp[i] = in[(((long long)b * 3 + c) * IMG + gy * P + py) * IMG + gx * P + px];
  }
}

__global__ void k_asm_main(const float* __restrict__ XE, const float* __restrict__ cls,
                           const float* __restrict__ pos, float* __restrict__ XQ,
                           float* __restrict__ XM) {
  const long long total = (long long)BATCH * NTOK * D;
  for (long long i = blockIdx.x * (long long)blockDim.x + threadIdx.x; i < total;
       i += (long long)gridDim.x * blockDim.x) {
    const int d = (int)(i % D);
    const int t = (int)((i / D) % NTOK);
    const int b = (int)(i / ((long long)D * NTOK));
    float v = (t == 0) ? cls[d] : XE[((long long)b * NPATCH + (t - 1)) * D + d];
    v += pos[(long long)t * D + d];
    XQ[i] = v;
    XM[i] = v;
  }
}

__global__ void k_asm_sub(const float* __restrict__ XE, const float* __restrict__ cls,
                          const float* __restrict__ pos, const float* __restrict__ subp,
                          float* __restrict__ XS) {
  const long long total = (long long)BATCH * NSUB * D;
  for (long long i = blockIdx.x * (long long)blockDim.x + threadIdx.x; i < total;
       i += (long long)gridDim.x * blockDim.x) {
    const int d = (int)(i % D);
    const int u = (int)((i / D) % NSUB);
    const int b = (int)(i / ((long long)D * NSUB));
    float v;
    if (u == 0) v = cls[d] + pos[d];
    else if (u <= PLEN) v = subp[(long long)(u - 1) * D + d] + pos[d];
    else {
      const int t = u - PLEN;  // 1..196
      v = XE[((long long)b * NPATCH + (t - 1)) * D + d] + pos[(long long)t * D + d];
    }
    XS[i] = v;
  }
}

// ----------------------- prompt key selection ------------------------------
__global__ __launch_bounds__(512) void k_select(const float* __restrict__ LNQ,
                                                const float* __restrict__ mkey,
                                                const float* __restrict__ skey,
                                                const int* __restrict__ tidp,
                                                int* __restrict__ selc,
                                                int* __restrict__ mask) {
  const int b = threadIdx.x >> 5, lane = threadIdx.x & 31;
  if (b >= BATCH) return;
  const int tid = tidp[0];
  const float* q = LNQ + (long long)b * NTOK * D;  // token 0 of LN'd sequence
  float ss = 0.f;
  for (int j = lane; j < D; j += 32) ss += q[j] * q[j];
  ss = wsum(ss);
  const float qn = rsqrtf(fmaxf(ss, 1e-12f));
  float best = -1e30f;
  int bi = 0;
  for (int c = 0; c <= tid + 1; ++c) {
    const float* kk = (c <= tid) ? (mkey + (long long)c * D) : skey;
    float cs = 0.f, dt = 0.f;
    for (int j = lane; j < D; j += 32) {
      const float kv = kk[j];
      cs += kv * kv;
      dt += q[j] * kv;
    }
    cs = wsum(cs);
    dt = wsum(dt);
    const float sim = dt * qn * rsqrtf(fmaxf(cs, 1e-12f));
    if (sim > best) { best = sim; bi = c; }   // first-max argmax
  }
  if (lane == 0) {
    mask[b] = (bi <= tid) ? 1 : 0;
    int sm = bi;
    if (sm > tid) sm = tid;
    if (sm > NTASK - 1) sm = NTASK - 1;
    selc[b] = sm;
  }
}

// --------------------------- heads / epilogue ------------------------------
__global__ void k_row0(const float* __restrict__ LN, float* __restrict__ V) {
  for (int i = blockIdx.x * blockDim.x + threadIdx.x; i < BATCH * D;
       i += gridDim.x * blockDim.x) {
    const int b = i / D, d = i % D;
    V[i] = LN[(long long)b * NTOK * D + d];
  }
}

__global__ void k_meanp(const float* __restrict__ LN, float* __restrict__ V) {
  for (int i = blockIdx.x * blockDim.x + threadIdx.x; i < BATCH * D;
       i += gridDim.x * blockDim.x) {
    const int b = i / D, d = i % D;
    float s = 0.f;
    for (int j = 1; j <= PLEN; ++j) s += LN[((long long)b * NSUB + j) * D + d];
    V[i] = s * (1.f / PLEN);
  }
}

__global__ __launch_bounds__(32) void k_classify(const float* __restrict__ vecs,
                                                 const float* __restrict__ fw,
                                                 const float* __restrict__ fb,
                                                 float* __restrict__ logit) {
  const int c = blockIdx.x, b = blockIdx.y, lane = threadIdx.x & 31;
  const float* v = vecs + (long long)b * D;
  const float* w = fw + (long long)c * D;
  float s = 0.f;
  for (int j = lane; j < D; j += 32) s += v[j] * w[j];
  s = wsum(s);
  if (lane == 0) logit[b * NCLS + c] = s + fb[c];
}

__global__ void k_final(const float* __restrict__ Ml, const float* __restrict__ Sl,
                        const int* __restrict__ mask, float* __restrict__ out) {
  for (int i = blockIdx.x * blockDim.x + threadIdx.x; i < BATCH * NCLS;
       i += gridDim.x * blockDim.x) {
    const int b = i / NCLS;
    out[i] = mask[b] ? Ml[i] : Sl[i];
  }
}

}  // namespace vitp

// ===========================================================================
extern "C" void kernel_launch(void* const* d_in, const int* in_sizes, int n_in,
                              void* d_out, int out_size, void* d_ws, size_t ws_size,
                              hipStream_t stream) {
  using namespace vitp;
  (void)in_sizes; (void)n_in; (void)out_size; (void)ws_size;

  const float* inputs   = (const float*)d_in[0];
  const int*   task_id  = (const int*)d_in[1];
  const float* patch_w  = (const float*)d_in[2];
  const float* patch_b  = (const float*)d_in[3];
  const float* cls_tok  = (const float*)d_in[4];
  const float* pos      = (const float*)d_in[5];
  const float* qkv_w    = (const float*)d_in[6];
  const float* qkv_b    = (const float*)d_in[7];
  const float* proj_w   = (const float*)d_in[8];
  const float* proj_b   = (const float*)d_in[9];
  const float* ln1_w    = (const float*)d_in[10];
  const float* ln1_b    = (const float*)d_in[11];
  const float* ln2_w    = (const float*)d_in[12];
  const float* ln2_b    = (const float*)d_in[13];
  const float* fc1_w    = (const float*)d_in[14];
  const float* fc1_b    = (const float*)d_in[15];
  const float* fc2_w    = (const float*)d_in[16];
  const float* fc2_b    = (const float*)d_in[17];
  const float* norm_w   = (const float*)d_in[18];
  const float* norm_b   = (const float*)d_in[19];
  const float* fc_w     = (const float*)d_in[20];
  const float* fc_b     = (const float*)d_in[21];
  const float* main_key = (const float*)d_in[22];
  const float* sub_key  = (const float*)d_in[23];
  const float* main_pr  = (const float*)d_in[24];
  const float* sub_pr   = (const float*)d_in[25];
  float* out = (float*)d_out;

  // ---------------- workspace arena (floats) ----------------
  float* W0 = (float*)d_ws;
  size_t off = 0;
  auto take = [&](size_t n) { float* p = W0 + off; off += n; return p; };
  float* Xp    = take((size_t)BATCH * NPATCH * D);
  float* XE    = take((size_t)BATCH * NPATCH * D);
  float* XQ    = take((size_t)BATCH * NTOK * D);
  float* XM    = take((size_t)BATCH * NTOK * D);
  float* XS    = take((size_t)BATCH * NSUB * D);
  float* LN    = take((size_t)BATCH * NSUB * D);
  float* QKV   = take((size_t)BATCH * NSUB * 3 * D);
  float* ATT   = take((size_t)BATCH * NSUB * D);
  float* HH    = take((size_t)BATCH * NSUB * 4 * D);
  float* TT    = take((size_t)BATCH * NSUB * D);
  float* CLSV  = take((size_t)BATCH * D);
  float* MEANV = take((size_t)BATCH * D);
  float* MLOG  = take((size_t)BATCH * NCLS);
  float* SLOG  = take((size_t)BATCH * NCLS);
  int*   SELC  = (int*)take(64);
  int*   MASK  = SELC + 16;
  // deep-attention scratch aliased inside HH (disjoint lifetimes on the stream)
  float* KP = HH;
  float* VP = KP + (size_t)BATCH * KPAD * D;
  float* SB = VP + (size_t)BATCH * KPAD * D;
  float* AV = SB + (size_t)BATCH * MPAD * KPAD;

  auto gemm = [&](const float* A, const float* Wm, const float* bias, float* C,
                  int M, int N, int K, int lda, int ldw, int ldc,
                  long long sA, long long sW, long long sC, int nb, int tb, int act) {
    dim3 g((N + 31) / 32, (M + 63) / 64, nb);
    if (tb == 0) {
      if (act == 0)
        k_gemm<0, 0><<<g, 128, 0, stream>>>(A, Wm, bias, C, M, N, K, lda, ldw, ldc, sA, sW, sC);
      else
        k_gemm<0, 1><<<g, 128, 0, stream>>>(A, Wm, bias, C, M, N, K, lda, ldw, ldc, sA, sW, sC);
    } else {
      k_gemm<1, 0><<<g, 128, 0, stream>>>(A, Wm, bias, C, M, N, K, lda, ldw, ldc, sA, sW, sC);
    }
  };
  auto ln = [&](const float* x, const float* w, const float* b, float* y, int rows) {
    k_ln<<<(rows + 7) / 8, 256, 0, stream>>>(x, w, b, y, rows);
  };
  auto add = [&](float* y, const float* t, long long n) {
    long long n4 = n >> 2;
    int g = (int)((n4 + 255) / 256); if (g > 4096) g = 4096;
    k_add<<<g, 256, 0, stream>>>(y, t, n4);
  };

  auto block_std = [&](float* X, int Nn, int i) {
    const int rows = BATCH * Nn;
    ln(X, ln1_w + (size_t)i * D, ln1_b + (size_t)i * D, LN, rows);
    gemm(LN, qkv_w + (size_t)i * 3 * D * D, qkv_b + (size_t)i * 3 * D, QKV,
         rows, 3 * D, D, D, D, 3 * D, 0, 0, 0, 1, 0, 0);
    k_mha<<<dim3(Nn, HEADS, BATCH), 32, 0, stream>>>(QKV, ATT, Nn);
    gemm(ATT, proj_w + (size_t)i * D * D, proj_b + (size_t)i * D, TT,
         rows, D, D, D, D, D, 0, 0, 0, 1, 0, 0);
    add(X, TT, (long long)rows * D);
    ln(X, ln2_w + (size_t)i * D, ln2_b + (size_t)i * D, LN, rows);
    gemm(LN, fc1_w + (size_t)i * 4 * D * D, fc1_b + (size_t)i * 4 * D, HH,
         rows, 4 * D, D, D, D, 4 * D, 0, 0, 0, 1, 0, 1);
    gemm(HH, fc2_w + (size_t)i * D * 4 * D, fc2_b + (size_t)i * D, TT,
         rows, D, 4 * D, 4 * D, 4 * D, D, 0, 0, 0, 1, 0, 0);
    add(X, TT, (long long)rows * D);
  };

  auto block_deep = [&](float* X, int i, int li) {
    const int rows = BATCH * NTOK;
    ln(X, ln1_w + (size_t)i * D, ln1_b + (size_t)i * D, LN, rows);
    gemm(LN, qkv_w + (size_t)i * 3 * D * D, qkv_b + (size_t)i * 3 * D, QKV,
         rows, 3 * D, D, D, D, 3 * D, 0, 0, 0, 1, 0, 0);
    k_buildkv<<<dim3(KPAD, BATCH), 256, 0, stream>>>(QKV, main_pr, SELC, KP, VP, li);
    // S = Q @ Kp^T  (batched, K over full D, cols padded to 224)
    gemm(QKV, KP, nullptr, SB, NTOK, KPAD, D, 3 * D, D, KPAD,
         (long long)NTOK * 3 * D, (long long)KPAD * D, (long long)MPAD * KPAD,
         BATCH, 0, 0);
    k_softmax<<<dim3(NTOK, BATCH), 32, 0, stream>>>(SB);
    // AV = P @ Vp   (batched, TRANSB: V is (K,N) row-major)
    gemm(SB, VP, nullptr, AV, NTOK, D, KPAD, KPAD, D, D,
         (long long)MPAD * KPAD, (long long)KPAD * D, (long long)NTOK * D,
         BATCH, 1, 0);
    k_avremap<<<2048, 256, 0, stream>>>(AV, ATT);
    gemm(ATT, proj_w + (size_t)i * D * D, proj_b + (size_t)i * D, TT,
         rows, D, D, D, D, D, 0, 0, 0, 1, 0, 0);
    add(X, TT, (long long)rows * D);
    ln(X, ln2_w + (size_t)i * D, ln2_b + (size_t)i * D, LN, rows);
    gemm(LN, fc1_w + (size_t)i * 4 * D * D, fc1_b + (size_t)i * 4 * D, HH,
         rows, 4 * D, D, D, D, 4 * D, 0, 0, 0, 1, 0, 1);
    gemm(HH, fc2_w + (size_t)i * D * 4 * D, fc2_b + (size_t)i * D, TT,
         rows, D, 4 * D, 4 * D, 4 * D, D, 0, 0, 0, 1, 0, 0);
    add(X, TT, (long long)rows * D);
  };

  // ---------------- patch embed ----------------
  k_patch<<<2048, 256, 0, stream>>>(inputs, Xp);
  gemm(Xp, patch_w, patch_b, XE, BATCH * NPATCH, D, D, D, D, D, 0, 0, 0, 1, 0, 0);
  k_asm_main<<<2048, 256, 0, stream>>>(XE, cls_tok, pos, XQ, XM);
  k_asm_sub<<<2048, 256, 0, stream>>>(XE, cls_tok, pos, sub_pr, XS);

  // ---------------- query pass + selection ----------------
  for (int i = 0; i < DEPTH; ++i) block_std(XQ, NTOK, i);
  ln(XQ, norm_w, norm_b, LN, BATCH * NTOK);
  k_select<<<1, 512, 0, stream>>>(LN, main_key, sub_key, task_id, SELC, MASK);

  // ---------------- main branch (deep prompts at layers 2,3,4) -------------
  for (int i = 0; i < DEPTH; ++i) {
    if (i >= 2 && i <= 4) block_deep(XM, i, i - 2);
    else block_std(XM, NTOK, i);
  }
  ln(XM, norm_w, norm_b, LN, BATCH * NTOK);
  k_row0<<<64, 256, 0, stream>>>(LN, CLSV);
  k_classify<<<dim3(NCLS, BATCH), 32, 0, stream>>>(CLSV, fc_w, fc_b, MLOG);

  // ---------------- sub branch (5 prompt tokens inserted) ------------------
  for (int i = 0; i < DEPTH; ++i) block_std(XS, NSUB, i);
  ln(XS, norm_w, norm_b, LN, BATCH * NSUB);
  k_meanp<<<64, 256, 0, stream>>>(LN, MEANV);
  k_classify<<<dim3(NCLS, BATCH), 32, 0, stream>>>(MEANV, fc_w, fc_b, SLOG);

  // ---------------- scatter by mask ----------------
  k_final<<<8, 256, 0, stream>>>(MLOG, SLOG, MASK, out);
}
